// SR_GNN_att_agg_27384711479641
// MI455X (gfx1250) — compile-verified, matrix operand order checked by reading, not run on hardware
//
#include <hip/hip_runtime.h>
#include <hip/hip_bf16.h>

typedef __bf16 bf16_t;
typedef __attribute__((ext_vector_type(16))) __bf16 v16bf;
typedef __attribute__((ext_vector_type(8)))  float  v8f;

#define NNODES 40960
#define NSESS  4096
#define NEDGES 81920
#define DEMB   64
#define HDIM   100
#define FIN    321
#define KPX    352   /* 321 padded to multiple of 32 */
#define LDH    128   /* 100 padded to multiple of 32 */
#define LD3    304   /* 300 padded to multiple of 16 */
#define VITEMS 50000
#define NTILES_FC 3125        /* 50000/16 */
#define NT_PER_WAVE 5         /* N-tiles per wave in FC GEMM */
#define NT_PER_BLOCK 40       /* 8 waves * 5 tiles */
#define NGROUPS_FC 79         /* ceil(3125/40) */

static const int TPB = 256;

__device__ __forceinline__ float sigmoidf_(float x) { return 1.f / (1.f + __expf(-x)); }

// ---------------- utility kernels ----------------
__global__ void k_fill_f32(float* p, float v, long n) {
    long i = blockIdx.x * (long)blockDim.x + threadIdx.x;
    if (i < n) p[i] = v;
}
__global__ void k_fill_i32(int* p, int v, long n) {
    long i = blockIdx.x * (long)blockDim.x + threadIdx.x;
    if (i < n) p[i] = v;
}
__global__ void k_f32_to_bf16(const float* __restrict__ s, bf16_t* __restrict__ d, long n) {
    long i = blockIdx.x * (long)blockDim.x + threadIdx.x;
    if (i < n) d[i] = (bf16_t)s[i];
}

// Convert fp32 weight W[Kc,Nc] (row-major) into bf16 transposed+padded Bt[Npt,Kp]
__global__ void k_cvt_wt(const float* __restrict__ W, bf16_t* __restrict__ Bt,
                         int Kc, int Nc, int Kp, int Npt) {
    long idx = blockIdx.x * (long)blockDim.x + threadIdx.x;
    long total = (long)Npt * Kp;
    if (idx >= total) return;
    int n = (int)(idx / Kp), k = (int)(idx % Kp);
    float v = (n < Nc && k < Kc) ? W[(size_t)k * Nc + n] : 0.f;
    Bt[idx] = (bf16_t)v;
}

// ---------------- node embedding gather -> x bf16 [N, KPX] ----------------
__global__ void k_embed(const float* __restrict__ price,
                        const int* __restrict__ cat, const int* __restrict__ sub,
                        const int* __restrict__ elem, const int* __restrict__ brand,
                        const int* __restrict__ item,
                        const float* __restrict__ ecat, const float* __restrict__ esub,
                        const float* __restrict__ eelem, const float* __restrict__ ebrand,
                        const float* __restrict__ eitem,
                        bf16_t* __restrict__ xb) {
    long idx = blockIdx.x * (long)blockDim.x + threadIdx.x;
    long total = (long)NNODES * KPX;
    if (idx >= total) return;
    int node = (int)(idx / KPX);
    int c = (int)(idx % KPX);
    float v = 0.f;
    if (c == 0) {
        v = price[node];
    } else if (c < FIN) {
        int cc = c - 1;
        int seg = cc >> 6;
        int d = cc & 63;
        const float* tab; int row;
        switch (seg) {
            case 0:  tab = ecat;   row = cat[node];   break;
            case 1:  tab = esub;   row = sub[node];   break;
            case 2:  tab = eelem;  row = elem[node];  break;
            case 3:  tab = ebrand; row = brand[node]; break;
            default: tab = eitem;  row = item[node];  break;
        }
        v = tab[(size_t)row * DEMB + d];
    }
    xb[idx] = (bf16_t)v;
}

// ---------------- generic bf16 WMMA GEMM: C[M,N] = A[M,K] * Bt[N,K]^T + bias ------
union Frag16 { v16bf v; unsigned u[8]; };

__global__ void __launch_bounds__(256) k_gemm_bf16(
        const bf16_t* __restrict__ A, int lda,
        const bf16_t* __restrict__ Bt, int ldb,
        const float* __restrict__ bias,
        float* __restrict__ C, int ldc,
        int Mtiles, int Ntiles, int K, int Ncols, int Nstore, int relu) {
    int flat = blockIdx.x * blockDim.x + threadIdx.x;
    int wid = flat >> 5;                 // wave32
    if (wid >= Mtiles * Ntiles) return;  // wave-uniform exit (EXEC all-1 inside WMMA)
    int lane = threadIdx.x & 31;
    int tm = wid / Ntiles, tn = wid % Ntiles;
    int half = lane >> 4;
    int sub  = lane & 15;
    const bf16_t* arow = A  + (size_t)(tm * 16 + sub) * lda;
    const bf16_t* brow = Bt + (size_t)(tn * 16 + sub) * ldb;
    int off = half * 4;
    v8f acc = {0.f, 0.f, 0.f, 0.f, 0.f, 0.f, 0.f, 0.f};
    for (int k0 = 0; k0 < K; k0 += 32) {
        Frag16 fa, fb;
        const unsigned* pa = (const unsigned*)(arow + k0);
        const unsigned* pb = (const unsigned*)(brow + k0);
#pragma unroll
        for (int j = 0; j < 4; ++j) {
            fa.u[j]     = pa[off + j];
            fa.u[4 + j] = pa[8 + off + j];
            fb.u[j]     = pb[off + j];
            fb.u[4 + j] = pb[8 + off + j];
        }
        acc = __builtin_amdgcn_wmma_f32_16x16x32_bf16(
                false, fa.v, false, fb.v, (short)0, acc, false, false);
    }
    int n = tn * 16 + sub;
    if (n >= Nstore) return;
    float badd = (bias != nullptr && n < Ncols) ? bias[n] : 0.f;
#pragma unroll
    for (int r = 0; r < 8; ++r) {
        int m = tm * 16 + half * 8 + r;
        float v = acc[r] + badd;
        if (relu) v = fmaxf(v, 0.f);
        if (n >= Ncols) v = 0.f;
        C[(size_t)m * ldc + n] = v;
    }
}

// ---------------- FC GEMM (dominant): LDS-staged A via async copy, 5 N-tiles/wave --
// Block: 256 threads = 8 waves, all sharing one M-tile of A (16x128 bf16 = 4KB in LDS).
// A is brought in with GLOBAL_LOAD_ASYNC_TO_LDS_B128 (ASYNCcnt) — one 16B chunk per
// lane — then each wave reads its A fragments from LDS (ds_load_b128) and streams B
// fragments from L2, issuing 5 WMMAs per A fragment. Output stores are the HBM
// roofline (819 MB), so everything else just has to stay off the critical path.
__global__ void __launch_bounds__(256) k_gemm_fc(
        const bf16_t* __restrict__ A,      /* pooled bf16 [NSESS, LDH] */
        const bf16_t* __restrict__ Bt,     /* Wfc_t bf16 [VITEMS, LDH] */
        const float* __restrict__ bias,    /* b_fc [VITEMS] */
        float* __restrict__ C) {           /* out [NSESS, VITEMS] */
    __shared__ bf16_t sA[16 * LDH];        // 4 KB

    int tid  = threadIdx.x;
    int lane = tid & 31;
    int wave = tid >> 5;
    int tm = blockIdx.x / NGROUPS_FC;      // 0..255
    int g  = blockIdx.x % NGROUPS_FC;      // 0..78
    int tnbase = g * NT_PER_BLOCK + wave * NT_PER_WAVE;

    // --- async stage A tile into LDS: thread t copies bytes [16t, 16t+16) ---
    {
        const bf16_t* gsrc = A + (size_t)tm * 16 * LDH + (size_t)tid * 8; // 8 bf16 = 16B
        unsigned lds_lo = (unsigned)(size_t)(&sA[0]) + (unsigned)(tid * 16);
        asm volatile("global_load_async_to_lds_b128 %0, %1, off"
                     :: "v"(lds_lo), "v"(gsrc) : "memory");
        asm volatile("s_wait_asynccnt 0x0" ::: "memory");
    }
    __syncthreads();

    int half = lane >> 4;
    int sub  = lane & 15;
    int off  = half * 4;
    const bf16_t* arow = &sA[sub * LDH];

    v8f acc[NT_PER_WAVE];
#pragma unroll
    for (int t = 0; t < NT_PER_WAVE; ++t)
        acc[t] = (v8f){0.f, 0.f, 0.f, 0.f, 0.f, 0.f, 0.f, 0.f};

#pragma unroll
    for (int k0 = 0; k0 < LDH; k0 += 32) {
        Frag16 fa;
        const unsigned* pa = (const unsigned*)(arow + k0);
#pragma unroll
        for (int j = 0; j < 4; ++j) { fa.u[j] = pa[off + j]; fa.u[4 + j] = pa[8 + off + j]; }
#pragma unroll
        for (int t = 0; t < NT_PER_WAVE; ++t) {
            int tn = tnbase + t;
            int tnl = tn < NTILES_FC ? tn : NTILES_FC - 1;  // clamp loads in-bounds
            const unsigned* pb =
                (const unsigned*)(Bt + (size_t)(tnl * 16 + sub) * LDH + k0);
            Frag16 fb;
#pragma unroll
            for (int j = 0; j < 4; ++j) { fb.u[j] = pb[off + j]; fb.u[4 + j] = pb[8 + off + j]; }
            acc[t] = __builtin_amdgcn_wmma_f32_16x16x32_bf16(
                    false, fa.v, false, fb.v, (short)0, acc[t], false, false);
        }
    }

#pragma unroll
    for (int t = 0; t < NT_PER_WAVE; ++t) {
        int n = (tnbase + t) * 16 + sub;
        if (n >= VITEMS) continue;
        float badd = bias[n];
#pragma unroll
        for (int r = 0; r < 8; ++r) {
            int m = tm * 16 + half * 8 + r;
            C[(size_t)m * VITEMS + n] = acc[t][r] + badd;
        }
    }
}

// ---------------- graph message passing ----------------
__global__ void k_degree(const int* __restrict__ ei, float* __restrict__ deg) {
    long e = blockIdx.x * (long)blockDim.x + threadIdx.x;
    if (e < NEDGES) atomicAdd(&deg[ei[NEDGES + e]], 1.f);
}
__global__ void k_invdeg(const float* __restrict__ deg, float* __restrict__ inv) {
    long i = blockIdx.x * (long)blockDim.x + threadIdx.x;
    if (i < NNODES) { float d = deg[i]; inv[i] = d > 0.f ? 1.f / d : 0.f; }
}
__global__ void k_scatter(const int* __restrict__ ei, const float* __restrict__ h,
                          float* __restrict__ msg) {
    long idx = blockIdx.x * (long)blockDim.x + threadIdx.x;
    long total = (long)NEDGES * HDIM;
    if (idx >= total) return;
    int e = (int)(idx / HDIM), f = (int)(idx % HDIM);
    int s = ei[e], d = ei[NEDGES + e];
    atomicAdd(&msg[(size_t)d * LDH + f], h[(size_t)s * LDH + f]);
}
__global__ void k_scale_msg(float* __restrict__ msg, bf16_t* __restrict__ msgb,
                            const float* __restrict__ inv) {
    long idx = blockIdx.x * (long)blockDim.x + threadIdx.x;
    long total = (long)NNODES * LDH;
    if (idx >= total) return;
    int node = (int)(idx / LDH);
    float v = msg[idx] * inv[node];
    msg[idx] = v;
    msgb[idx] = (bf16_t)v;
}

// ---------------- GRU cell (gates [r,z,n]) ----------------
__global__ void k_gru(const float* __restrict__ gi, const float* __restrict__ gh,
                      float* __restrict__ h, bf16_t* __restrict__ hb) {
    long idx = blockIdx.x * (long)blockDim.x + threadIdx.x;
    long total = (long)NNODES * HDIM;
    if (idx >= total) return;
    int node = (int)(idx / HDIM), f = (int)(idx % HDIM);
    size_t b3 = (size_t)node * LD3;
    float ir = gi[b3 + f], iz = gi[b3 + HDIM + f], inn = gi[b3 + 2 * HDIM + f];
    float hr = gh[b3 + f], hz = gh[b3 + HDIM + f], hn  = gh[b3 + 2 * HDIM + f];
    float r = sigmoidf_(ir + hr);
    float z = sigmoidf_(iz + hz);
    float nn = tanhf(inn + r * hn);
    size_t hi = (size_t)node * LDH + f;
    float out = (1.f - z) * nn + z * h[hi];
    h[hi] = out;
    hb[hi] = (bf16_t)out;
}

// ---------------- session bounds / last node ----------------
__global__ void k_bounds(const int* __restrict__ batch, int* __restrict__ st, int* __restrict__ en) {
    long i = blockIdx.x * (long)blockDim.x + threadIdx.x;
    if (i >= NNODES) return;
    int b = batch[i];
    atomicMin(&st[b], (int)i);
    atomicMax(&en[b], (int)i);
}
__global__ void k_gather_last(const bf16_t* __restrict__ xb, const int* __restrict__ lastidx,
                              bf16_t* __restrict__ xl) {
    long idx = blockIdx.x * (long)blockDim.x + threadIdx.x;
    long total = (long)NSESS * KPX;
    if (idx >= total) return;
    int s = (int)(idx / KPX), k = (int)(idx % KPX);
    int li = lastidx[s]; if (li < 0) li = 0;
    xl[idx] = xb[(size_t)li * KPX + k];
}
__global__ void k_addlast(float* __restrict__ h, bf16_t* __restrict__ hb,
                          const float* __restrict__ last, const int* __restrict__ batch) {
    long idx = blockIdx.x * (long)blockDim.x + threadIdx.x;
    long total = (long)NNODES * LDH;
    if (idx >= total) return;
    int node = (int)(idx / LDH), f = (int)(idx % LDH);
    float v = h[idx] + last[(size_t)batch[node] * LDH + f];
    h[idx] = v;
    hb[idx] = (bf16_t)v;
}

// ---------------- per-session softmax + weighted pool (LDS staging) ----------------
__global__ void k_pool(const float* __restrict__ gate, const float* __restrict__ h,
                       const int* __restrict__ st_, const int* __restrict__ en_,
                       float* __restrict__ pooled, bf16_t* __restrict__ pooledb) {
    __shared__ float s_alpha[64];
    int s = blockIdx.x;
    int st = st_[s], en = en_[s];
    int cnt = (en >= st) ? (en - st + 1) : 0;
    if (cnt > 64) cnt = 64;
    if (threadIdx.x == 0) {
        float gm = -3.4e38f;
        for (int i = 0; i < cnt; ++i) gm = fmaxf(gm, gate[st + i]);
        float ws = 0.f;
        for (int i = 0; i < cnt; ++i) { float w = __expf(gate[st + i] - gm); s_alpha[i] = w; ws += w; }
        float inv = ws > 0.f ? 1.f / ws : 0.f;
        for (int i = 0; i < cnt; ++i) s_alpha[i] *= inv;
    }
    __syncthreads();
    for (int f = threadIdx.x; f < LDH; f += blockDim.x) {
        float acc = 0.f;
        for (int i = 0; i < cnt; ++i) acc += s_alpha[i] * h[(size_t)(st + i) * LDH + f];
        pooled[(size_t)s * LDH + f] = acc;
        pooledb[(size_t)s * LDH + f] = (bf16_t)acc;
    }
}

// ---------------- host orchestration ----------------
extern "C" void kernel_launch(void* const* d_in, const int* in_sizes, int n_in,
                              void* d_out, int out_size, void* d_ws, size_t ws_size,
                              hipStream_t stream) {
    (void)in_sizes; (void)n_in; (void)out_size; (void)ws_size;
    const float* price   = (const float*)d_in[0];
    const int*   cat     = (const int*)d_in[1];
    const int*   sub     = (const int*)d_in[2];
    const int*   elem    = (const int*)d_in[3];
    const int*   brand   = (const int*)d_in[4];
    const int*   item    = (const int*)d_in[5];
    const int*   ei      = (const int*)d_in[6];
    const int*   batch   = (const int*)d_in[7];
    const float* ecat    = (const float*)d_in[8];
    const float* esub    = (const float*)d_in[9];
    const float* eelem   = (const float*)d_in[10];
    const float* ebrand  = (const float*)d_in[11];
    const float* eitem   = (const float*)d_in[12];
    const float* W_msg   = (const float*)d_in[13];
    const float* b_msg   = (const float*)d_in[14];
    const float* W_ih    = (const float*)d_in[15];
    const float* W_hh    = (const float*)d_in[16];
    const float* b_ih    = (const float*)d_in[17];
    const float* b_hh    = (const float*)d_in[18];
    const float* W_last  = (const float*)d_in[19];
    const float* b_last  = (const float*)d_in[20];
    const float* W_g1    = (const float*)d_in[21];
    const float* b_g1    = (const float*)d_in[22];
    const float* W_g2    = (const float*)d_in[23];
    const float* b_g2    = (const float*)d_in[24];
    const float* W_fc    = (const float*)d_in[25];
    const float* b_fc    = (const float*)d_in[26];
    float* out = (float*)d_out;

    // workspace carve (~245 MB total)
    char* p = (char*)d_ws;
    auto alloc = [&](size_t bytes) -> void* {
        void* r = (void*)p;
        p += (bytes + 255) & ~(size_t)255;
        return r;
    };
    bf16_t* xb       = (bf16_t*)alloc((size_t)NNODES * KPX * 2);
    bf16_t* xlast    = (bf16_t*)alloc((size_t)NSESS * KPX * 2);
    bf16_t* Wmsg_t   = (bf16_t*)alloc((size_t)128 * KPX * 2);
    bf16_t* Wih_t    = (bf16_t*)alloc((size_t)LD3 * LDH * 2);
    bf16_t* Whh_t    = (bf16_t*)alloc((size_t)LD3 * LDH * 2);
    bf16_t* Wlast_t  = (bf16_t*)alloc((size_t)128 * KPX * 2);
    bf16_t* Wg1_t    = (bf16_t*)alloc((size_t)128 * LDH * 2);
    bf16_t* Wg2_t    = (bf16_t*)alloc((size_t)16 * LDH * 2);
    bf16_t* Wfc_t    = (bf16_t*)alloc((size_t)VITEMS * LDH * 2);
    float*  hbuf     = (float*)alloc((size_t)NNODES * LDH * 4);
    bf16_t* hbf      = (bf16_t*)alloc((size_t)NNODES * LDH * 2);
    float*  msg      = (float*)alloc((size_t)NNODES * LDH * 4);
    bf16_t* msgbf    = (bf16_t*)alloc((size_t)NNODES * LDH * 2);
    float*  gi       = (float*)alloc((size_t)NNODES * LD3 * 4);
    float*  gh       = (float*)alloc((size_t)NNODES * LD3 * 4);
    float*  deg      = (float*)alloc((size_t)NNODES * 4);
    float*  invd     = (float*)alloc((size_t)NNODES * 4);
    int*    segst    = (int*)alloc((size_t)NSESS * 4);
    int*    segen    = (int*)alloc((size_t)NSESS * 4);
    float*  lastf    = (float*)alloc((size_t)NSESS * LDH * 4);
    float*  g1out    = (float*)alloc((size_t)NNODES * LDH * 4);
    bf16_t* g1bf     = (bf16_t*)alloc((size_t)NNODES * LDH * 2);
    float*  gatev    = (float*)alloc((size_t)NNODES * 4);
    float*  pooled   = (float*)alloc((size_t)NSESS * LDH * 4);
    bf16_t* pooledbf = (bf16_t*)alloc((size_t)NSESS * LDH * 2);

    auto nb = [](long n) { return (int)((n + TPB - 1) / TPB); };
    auto gemm_blocks = [](int mt, int nt) { return (mt * nt * 32 + TPB - 1) / TPB; };

    // 1) weights -> bf16 transposed+padded
    k_cvt_wt<<<nb((long)128 * KPX),  TPB, 0, stream>>>(W_msg,  Wmsg_t,  FIN,  HDIM,   KPX, 128);
    k_cvt_wt<<<nb((long)LD3 * LDH),  TPB, 0, stream>>>(W_ih,   Wih_t,   HDIM, 3*HDIM, LDH, LD3);
    k_cvt_wt<<<nb((long)LD3 * LDH),  TPB, 0, stream>>>(W_hh,   Whh_t,   HDIM, 3*HDIM, LDH, LD3);
    k_cvt_wt<<<nb((long)128 * KPX),  TPB, 0, stream>>>(W_last, Wlast_t, FIN,  HDIM,   KPX, 128);
    k_cvt_wt<<<nb((long)128 * LDH),  TPB, 0, stream>>>(W_g1,   Wg1_t,   HDIM, HDIM,   LDH, 128);
    k_cvt_wt<<<nb((long)16 * LDH),   TPB, 0, stream>>>(W_g2,   Wg2_t,   HDIM, 1,      LDH, 16);
    k_cvt_wt<<<nb((long)VITEMS*LDH), TPB, 0, stream>>>(W_fc,   Wfc_t,   HDIM, VITEMS, LDH, VITEMS);

    // 2) node embedding gather
    k_embed<<<nb((long)NNODES * KPX), TPB, 0, stream>>>(price, cat, sub, elem, brand, item,
                                                        ecat, esub, eelem, ebrand, eitem, xb);

    // 3) h = x @ W_msg + b_msg   (WMMA)
    k_gemm_bf16<<<gemm_blocks(NNODES/16, LDH/16), TPB, 0, stream>>>(
        xb, KPX, Wmsg_t, KPX, b_msg, hbuf, LDH, NNODES/16, LDH/16, KPX, HDIM, LDH, 0);
    k_f32_to_bf16<<<nb((long)NNODES * LDH), TPB, 0, stream>>>(hbuf, hbf, (long)NNODES * LDH);

    // 4) degree + inverse
    k_fill_f32<<<nb(NNODES), TPB, 0, stream>>>(deg, 0.f, NNODES);
    k_degree<<<nb(NEDGES), TPB, 0, stream>>>(ei, deg);
    k_invdeg<<<nb(NNODES), TPB, 0, stream>>>(deg, invd);

    // 5) msg = segment_sum(h[src], dst) * inv_deg
    k_fill_f32<<<nb((long)NNODES * LDH), TPB, 0, stream>>>(msg, 0.f, (long)NNODES * LDH);
    k_scatter<<<nb((long)NEDGES * HDIM), TPB, 0, stream>>>(ei, hbuf, msg);
    k_scale_msg<<<nb((long)NNODES * LDH), TPB, 0, stream>>>(msg, msgbf, invd);

    // 6) GRU gate GEMMs (WMMA) + elementwise cell
    k_gemm_bf16<<<gemm_blocks(NNODES/16, LD3/16), TPB, 0, stream>>>(
        msgbf, LDH, Wih_t, LDH, b_ih, gi, LD3, NNODES/16, LD3/16, LDH, 3*HDIM, LD3, 0);
    k_gemm_bf16<<<gemm_blocks(NNODES/16, LD3/16), TPB, 0, stream>>>(
        hbf, LDH, Whh_t, LDH, b_hh, gh, LD3, NNODES/16, LD3/16, LDH, 3*HDIM, LD3, 0);
    k_gru<<<nb((long)NNODES * HDIM), TPB, 0, stream>>>(gi, gh, hbuf, hbf);

    // 7) session bounds (segen doubles as last_idx since batch is sorted)
    k_fill_i32<<<nb(NSESS), TPB, 0, stream>>>(segst, 0x7fffffff, NSESS);
    k_fill_i32<<<nb(NSESS), TPB, 0, stream>>>(segen, -1, NSESS);
    k_bounds<<<nb(NNODES), TPB, 0, stream>>>(batch, segst, segen);

    // 8) last = x[last_idx] @ W_last + b_last ; h += last[batch]
    k_gather_last<<<nb((long)NSESS * KPX), TPB, 0, stream>>>(xb, segen, xlast);
    k_gemm_bf16<<<gemm_blocks(NSESS/16, LDH/16), TPB, 0, stream>>>(
        xlast, KPX, Wlast_t, KPX, b_last, lastf, LDH, NSESS/16, LDH/16, KPX, HDIM, LDH, 0);
    k_addlast<<<nb((long)NNODES * LDH), TPB, 0, stream>>>(hbuf, hbf, lastf, batch);

    // 9) attention gate: relu(h@W_g1+b_g1) @ W_g2 + b_g2
    k_gemm_bf16<<<gemm_blocks(NNODES/16, LDH/16), TPB, 0, stream>>>(
        hbf, LDH, Wg1_t, LDH, b_g1, g1out, LDH, NNODES/16, LDH/16, LDH, HDIM, LDH, 1);
    k_f32_to_bf16<<<nb((long)NNODES * LDH), TPB, 0, stream>>>(g1out, g1bf, (long)NNODES * LDH);
    k_gemm_bf16<<<gemm_blocks(NNODES/16, 1), TPB, 0, stream>>>(
        g1bf, LDH, Wg2_t, LDH, b_g2, gatev, 1, NNODES/16, 1, LDH, 1, 1, 0);

    // 10) segment softmax + weighted pooling
    k_pool<<<NSESS, 128, 0, stream>>>(gatev, hbuf, segst, segen, pooled, pooledbf);

    // 11) final scoring: pooled @ W_fc + b_fc -> [4096, 50000]
    //     LDS-staged A (async copy), 5 N-tiles per wave, store-bound at HBM roofline.
    k_gemm_fc<<<256 * NGROUPS_FC, TPB, 0, stream>>>(pooledbf, Wfc_t, b_fc, out);
}
